// MAGNOEncoder_87651692577273
// MI455X (gfx1250) — matrix operand order repaired and checked by reading, not compiled, not run on hardware
//
#include <hip/hip_runtime.h>

typedef __attribute__((ext_vector_type(16))) _Float16 v16h;
typedef __attribute__((ext_vector_type(8)))  _Float16 v8h;
typedef __attribute__((ext_vector_type(2)))  _Float16 h2v;
typedef __attribute__((ext_vector_type(8)))  float    v8f;

#define B_   2
#define N_   2048
#define NL_  512
#define CD_  2
#define INC_ 16
#define C_   32
#define H_   32
#define R1SQ (0.07f * 0.07f)
#define R2SQ (0.14f * 0.14f)

#if __has_builtin(__builtin_amdgcn_wmma_f16_16x16x32_f16)
#define HAVE_WMMA_F16D 1
#else
#define HAVE_WMMA_F16D 0
#endif

// ---- hardware tanh if the gfx1250 builtin is declared -----------------------
__device__ __forceinline__ _Float16 tanh_h(_Float16 v) {
#if __has_builtin(__builtin_amdgcn_tanh_f16)
    return __builtin_amdgcn_tanh_f16(v);
#elif __has_builtin(__builtin_amdgcn_tanhh)
    return __builtin_amdgcn_tanhh(v);
#elif __has_builtin(__builtin_amdgcn_tanhf)
    return (_Float16)__builtin_amdgcn_tanhf((float)v);
#else
    // sigmoid identity: tanh(z) = 2/(1+exp(-2z)) - 1
    float z = (float)v;
    float s = __expf(-2.0f * z);
    return (_Float16)(2.0f * __builtin_amdgcn_rcpf(1.0f + s) - 1.0f);
#endif
}

// tanh-approx GELU on packed f16: 0.5*x*(1+tanh(0.79788456*(x+0.044715x^3)))
__device__ __forceinline__ h2v gelu_h2(h2v x) {
    h2v x3 = x * x * x;
    h2v z  = (_Float16)0.7978845608028654f * (x + (_Float16)0.044715f * x3);
    h2v t;
    t[0] = tanh_h(z[0]);
    t[1] = tanh_h(z[1]);
    h2v hx = (_Float16)0.5f * x;
    return hx * t + hx;
}

__device__ __forceinline__ float gelu_f32(float x) {
    float z = 0.7978845608028654f * fmaf(0.044715f * x * x, x, x);
#if __has_builtin(__builtin_amdgcn_tanhf)
    float t = __builtin_amdgcn_tanhf(z);
    return 0.5f * x * (1.0f + t);
#else
    return x * __builtin_amdgcn_rcpf(1.0f + __expf(-2.0f * z));
#endif
}

union V8HU  { v8h  v; h2v h[4]; unsigned u[4]; };
union V16HU { v16h v; unsigned u[8]; };

// ---------------------------------------------------------------------------
// Kernel 1: f = pndata @ W_lift + b_lift   (B,N,16)@(16,32) -> (B,N,32)
// ---------------------------------------------------------------------------
__global__ void lift_kernel(const float* __restrict__ pndata,
                            const float* __restrict__ Wl,
                            const float* __restrict__ bl,
                            float* __restrict__ f) {
    int idx = blockIdx.x * blockDim.x + threadIdx.x;  // over B*N*C
    if (idx >= B_ * N_ * C_) return;
    int c   = idx & (C_ - 1);
    int row = idx >> 5;  // b*N + j
    const float* p = pndata + (size_t)row * INC_;
    float acc = bl[c];
#pragma unroll
    for (int k = 0; k < INC_; ++k) acc = fmaf(p[k], Wl[k * C_ + c], acc);
    f[idx] = acc;
}

// ---------------------------------------------------------------------------
// Kernel 2: per (b, latent i): 3-layer kernel-MLP over all N points via WMMA,
// masked aggregation at two radii.
//
// Transposed formulation: D = A(W^T, contiguous-K A layout) x B(acts, one pair
// per lane column). Hidden layers use the f16-accumulating WMMA so D is
// already packed f16; gelu + an 8-dword xor-16 swap produce the next B
// operand with no conversions. Layer 3 accumulates in f32.
// ---------------------------------------------------------------------------
__launch_bounds__(128, 1)
__global__ void magno_kernel(const float* __restrict__ x_coord,
                             const float* __restrict__ latent,
                             const float* __restrict__ kW1, const float* __restrict__ kb1,
                             const float* __restrict__ kW2, const float* __restrict__ kb2,
                             const float* __restrict__ kW3, const float* __restrict__ kb3,
                             const float* __restrict__ f,
                             float* __restrict__ out) {
    __shared__ float red[2][C_];
    __shared__ float cnts[2];

    const int i    = blockIdx.x;   // latent token
    const int b    = blockIdx.y;   // batch
    const int tid  = threadIdx.x;
    const int lane = tid & 31;
    const int wave = tid >> 5;

    if (tid < C_) { red[0][tid] = 0.0f; red[1][tid] = 0.0f; }
    if (tid < 2)  cnts[tid] = 0.0f;
    __syncthreads();

    const int  col = lane & 15;
    const bool hi  = lane >= 16;
    const int  kb  = hi ? 8 : 0;   // channel (D row) base; A K-row base is 2*kb

    // ---- Loop-invariant A operands: W1^T, W2^T, W3^T (2 M-tiles each) ----
    // A k order used here (contiguous): lane m: halves 0..15 = K 0..15
    //                           lane m+16: halves 0..15 = K 16..31
    v16h aW1[2], aW2[2], aW3[2];
#pragma unroll
    for (int t = 0; t < 2; ++t) {
        const int cidx = col + 16 * t;  // hidden/output unit = A row M
#pragma unroll
        for (int h = 0; h < 16; ++h) aW1[t][h] = (_Float16)0.0f;
        if (!hi) {
#pragma unroll
            for (int k = 0; k < 2 * CD_; ++k) aW1[t][k] = (_Float16)kW1[k * H_ + cidx];
        }
#pragma unroll
        for (int h = 0; h < 8; ++h) {
            aW2[t][h]     = (_Float16)kW2[(2 * kb + h) * H_ + cidx];
            aW2[t][8 + h] = (_Float16)kW2[(2 * kb + 8 + h) * H_ + cidx];
            aW3[t][h]     = (_Float16)kW3[(2 * kb + h) * C_ + cidx];
            aW3[t][8 + h] = (_Float16)kW3[(2 * kb + 8 + h) * C_ + cidx];
        }
    }

    // ---- Bias tiles as WMMA C operand ----
    // f32 D layout: lane n, VGPR r -> M = r + kb.  f16 D layout: half h -> M = h + kb.
#if HAVE_WMMA_F16D
    v8h cbh1[2], cbh2[2];
#pragma unroll
    for (int t = 0; t < 2; ++t)
#pragma unroll
        for (int h = 0; h < 8; ++h) {
            cbh1[t][h] = (_Float16)kb1[16 * t + kb + h];
            cbh2[t][h] = (_Float16)kb2[16 * t + kb + h];
        }
#else
    v8f cb1[2], cb2[2];
#pragma unroll
    for (int t = 0; t < 2; ++t)
#pragma unroll
        for (int r = 0; r < 8; ++r) {
            cb1[t][r] = kb1[16 * t + kb + r];
            cb2[t][r] = kb2[16 * t + kb + r];
        }
#endif
    v8f cb3[2];
#pragma unroll
    for (int t = 0; t < 2; ++t)
#pragma unroll
        for (int r = 0; r < 8; ++r) cb3[t][r] = kb3[16 * t + kb + r];

    const float q0 = latent[(size_t)i * CD_ + 0];
    const float q1 = latent[(size_t)i * CD_ + 1];

    float acc1[16], acc2[16];
#pragma unroll
    for (int r = 0; r < 16; ++r) { acc1[r] = 0.0f; acc2[r] = 0.0f; }
    float c1 = 0.0f, c2 = 0.0f;

    const int NTILES = N_ / 16;  // 128
    for (int t = wave; t < NTILES; t += 4) {
        const int j = t * 16 + col;  // this lane's pair (point) index

        const float* xp = x_coord + ((size_t)b * N_ + j) * CD_;
        const float x0 = xp[0], x1 = xp[1];
        const float dx0 = x0 - q0, dx1 = x1 - q1;
        const float d2 = dx0 * dx0 + dx1 * dx1;
        const float m1 = (d2 <= R1SQ) ? 1.0f : 0.0f;
        const float m2 = (d2 <= R2SQ) ? 1.0f : 0.0f;

        // ---- B operand, layer 1 input: K = [y0,y1,q0,q1] padded to 32 ----
        // B k order used here (contiguous, same as A): lanes 0..15 hold K 0..15 (halves linear), lanes 16..31 K 16..31
        v16h bp;
#pragma unroll
        for (int h = 0; h < 16; ++h) bp[h] = (_Float16)0.0f;
        if (!hi) {
            bp[0] = (_Float16)x0; bp[1] = (_Float16)x1;
            bp[2] = (_Float16)q0; bp[3] = (_Float16)q1;
        }

#if HAVE_WMMA_F16D
        // ---- Layers 1 & 2 with f16-accumulating WMMA: D already packed f16 ----
        V8HU e0, e1;
        e0.v = __builtin_amdgcn_wmma_f16_16x16x32_f16(false, aW1[0], false, bp, (short)0, cbh1[0], false, false);
        e1.v = __builtin_amdgcn_wmma_f16_16x16x32_f16(false, aW1[1], false, bp, (short)0, cbh1[1], false, false);

        V16HU bh;
#pragma unroll
        for (int r = 0; r < 4; ++r) {
            h2v g0 = gelu_h2(e0.h[r]);
            h2v g1 = gelu_h2(e1.h[r]);
            unsigned u0, u1;
            __builtin_memcpy(&u0, &g0, 4);
            __builtin_memcpy(&u1, &g1, 4);
            unsigned o0 = __shfl_xor(u0, 16, 32);
            unsigned o1 = __shfl_xor(u1, 16, 32);
            bh.u[r]     = hi ? o1 : u0;   // K 0..7  (lo) / 16..23 (hi)
            bh.u[4 + r] = hi ? u1 : o0;   // K 8..15 (lo) / 24..31 (hi)
        }

        e0.v = __builtin_amdgcn_wmma_f16_16x16x32_f16(false, aW2[0], false, bh.v, (short)0, cbh2[0], false, false);
        e1.v = __builtin_amdgcn_wmma_f16_16x16x32_f16(false, aW2[1], false, bh.v, (short)0, cbh2[1], false, false);

        V16HU bh2;
#pragma unroll
        for (int r = 0; r < 4; ++r) {
            h2v g0 = gelu_h2(e0.h[r]);
            h2v g1 = gelu_h2(e1.h[r]);
            unsigned u0, u1;
            __builtin_memcpy(&u0, &g0, 4);
            __builtin_memcpy(&u1, &g1, 4);
            unsigned o0 = __shfl_xor(u0, 16, 32);
            unsigned o1 = __shfl_xor(u1, 16, 32);
            bh2.u[r]     = hi ? o1 : u0;
            bh2.u[4 + r] = hi ? u1 : o0;
        }
#else
        // ---- Fallback: f32-accumulating WMMA + pack-convert ----
        v8f d0f = __builtin_amdgcn_wmma_f32_16x16x32_f16(false, aW1[0], false, bp, (short)0, cb1[0], false, false);
        v8f d1f = __builtin_amdgcn_wmma_f32_16x16x32_f16(false, aW1[1], false, bp, (short)0, cb1[1], false, false);

        V16HU bh;
#pragma unroll
        for (int r = 0; r < 4; ++r) {
            h2v g0, g1;
            g0[0] = (_Float16)gelu_f32(d0f[2 * r]);
            g0[1] = (_Float16)gelu_f32(d0f[2 * r + 1]);
            g1[0] = (_Float16)gelu_f32(d1f[2 * r]);
            g1[1] = (_Float16)gelu_f32(d1f[2 * r + 1]);
            unsigned u0, u1;
            __builtin_memcpy(&u0, &g0, 4);
            __builtin_memcpy(&u1, &g1, 4);
            unsigned o0 = __shfl_xor(u0, 16, 32);
            unsigned o1 = __shfl_xor(u1, 16, 32);
            bh.u[r]     = hi ? o1 : u0;
            bh.u[4 + r] = hi ? u1 : o0;
        }

        d0f = __builtin_amdgcn_wmma_f32_16x16x32_f16(false, aW2[0], false, bh.v, (short)0, cb2[0], false, false);
        d1f = __builtin_amdgcn_wmma_f32_16x16x32_f16(false, aW2[1], false, bh.v, (short)0, cb2[1], false, false);

        V16HU bh2;
#pragma unroll
        for (int r = 0; r < 4; ++r) {
            h2v g0, g1;
            g0[0] = (_Float16)gelu_f32(d0f[2 * r]);
            g0[1] = (_Float16)gelu_f32(d0f[2 * r + 1]);
            g1[0] = (_Float16)gelu_f32(d1f[2 * r]);
            g1[1] = (_Float16)gelu_f32(d1f[2 * r + 1]);
            unsigned u0, u1;
            __builtin_memcpy(&u0, &g0, 4);
            __builtin_memcpy(&u1, &g1, 4);
            unsigned o0 = __shfl_xor(u0, 16, 32);
            unsigned o1 = __shfl_xor(u1, 16, 32);
            bh2.u[r]     = hi ? o1 : u0;
            bh2.u[4 + r] = hi ? u1 : o0;
        }
#endif

        // ---- Layer 3 (kvals), f32 accumulate for accurate aggregation ----
        v8f d0 = __builtin_amdgcn_wmma_f32_16x16x32_f16(false, aW3[0], false, bh2.v, (short)0, cb3[0], false, false);
        v8f d1 = __builtin_amdgcn_wmma_f32_16x16x32_f16(false, aW3[1], false, bh2.v, (short)0, cb3[1], false, false);

        // ---- masked aggregation: lane holds channels {kb..kb+7, 16+kb..+7} of pair j ----
        const float4* fp = (const float4*)(f + ((size_t)b * N_ + j) * C_);
        const int fi = kb >> 2;
        float4 fA = fp[fi], fB = fp[fi + 1], fC = fp[4 + fi], fD = fp[4 + fi + 1];
        float fv[16] = { fA.x, fA.y, fA.z, fA.w, fB.x, fB.y, fB.z, fB.w,
                         fC.x, fC.y, fC.z, fC.w, fD.x, fD.y, fD.z, fD.w };
#pragma unroll
        for (int r = 0; r < 8; ++r) {
            float tv0 = d0[r] * fv[r];
            float tv1 = d1[r] * fv[8 + r];
            acc1[r]     = fmaf(tv0, m1, acc1[r]);
            acc2[r]     = fmaf(tv0, m2, acc2[r]);
            acc1[8 + r] = fmaf(tv1, m1, acc1[8 + r]);
            acc2[8 + r] = fmaf(tv1, m2, acc2[8 + r]);
        }
        if (!hi) { c1 += m1; c2 += m2; }  // one count contribution per pair
    }

    // ---- reduce over the 16 pairs held in each 16-lane half ----
#pragma unroll
    for (int off = 1; off < 16; off <<= 1) {
#pragma unroll
        for (int r = 0; r < 16; ++r) {
            acc1[r] += __shfl_xor(acc1[r], off, 32);
            acc2[r] += __shfl_xor(acc2[r], off, 32);
        }
        c1 += __shfl_xor(c1, off, 32);
        c2 += __shfl_xor(c2, off, 32);
    }

    if (col == 0) {  // lanes 0 and 16 hold half-group totals
#pragma unroll
        for (int r = 0; r < 16; ++r) {
            int c = (r < 8) ? (kb + r) : (16 + kb + r - 8);
            atomicAdd(&red[0][c], acc1[r]);
            atomicAdd(&red[1][c], acc2[r]);
        }
        if (!hi) { atomicAdd(&cnts[0], c1); atomicAdd(&cnts[1], c2); }
    }
    __syncthreads();

    if (tid < C_) {
        float n1 = fmaxf(cnts[0], 1.0f);
        float n2 = fmaxf(cnts[1], 1.0f);
        out[((size_t)b * NL_ + i) * C_ + tid] = red[0][tid] / n1 + red[1][tid] / n2;
    }
}

// ---------------------------------------------------------------------------
extern "C" void kernel_launch(void* const* d_in, const int* in_sizes, int n_in,
                              void* d_out, int out_size, void* d_ws, size_t ws_size,
                              hipStream_t stream) {
    const float* x_coord = (const float*)d_in[0];
    const float* pndata  = (const float*)d_in[1];
    const float* latent  = (const float*)d_in[2];
    const float* W_lift  = (const float*)d_in[3];
    const float* b_lift  = (const float*)d_in[4];
    const float* kW1     = (const float*)d_in[5];
    const float* kb1     = (const float*)d_in[6];
    const float* kW2     = (const float*)d_in[7];
    const float* kb2     = (const float*)d_in[8];
    const float* kW3     = (const float*)d_in[9];
    const float* kb3     = (const float*)d_in[10];

    float* f   = (float*)d_ws;   // (B*N, 32) lifted features, 512 KB
    float* out = (float*)d_out;  // (B, NL, 32)

    const int total = B_ * N_ * C_;
    lift_kernel<<<(total + 255) / 256, 256, 0, stream>>>(pndata, W_lift, b_lift, f);

    dim3 grid(NL_, B_);
    magno_kernel<<<grid, 128, 0, stream>>>(x_coord, latent,
                                           kW1, kb1, kW2, kb2, kW3, kb3,
                                           f, out);
}
